// Attention_86131274154383
// MI455X (gfx1250) — matrix operand order repaired and checked by reading, not compiled, not run on hardware
//
#include <hip/hip_runtime.h>

// ---------------------------------------------------------------------------
// CDNA5 (gfx1250) attention block:
//   1) convert_weights : w_qkv/w_out f32 -> transposed bf16 hi/lo pairs
//   2) qkv_gemm        : X(8192x512) @ Wqkv(512x1536), split-bf16 WMMA,
//                        scatter to Q[b,h,n,d], K[b,h,n,d], Vt[b,h,d,n] (bf16)
//   3) attention       : flash-style per (b,h); K/V tiles staged block-wide
//                        into LDS with GLOBAL_LOAD_ASYNC_TO_LDS_B128
//                        (ASYNCcnt), WMMA bf16 for QK^T and P*V
//   4) out_gemm        : AO(8192x512)bf16 @ Wout(512x512) split-bf16 + bias
// ---------------------------------------------------------------------------

typedef __bf16 bf16;
typedef __attribute__((ext_vector_type(16))) __bf16 v16bf;
typedef __attribute__((ext_vector_type(8)))  __bf16 v8bf;
typedef __attribute__((ext_vector_type(8)))  float  v8f;
typedef __attribute__((ext_vector_type(4)))  int    v4i;

union V16 { v16bf v; v8bf h[2]; bf16 e[16]; };

#define WMMA_BF16(A, B, C) \
  __builtin_amdgcn_wmma_f32_16x16x32_bf16(false, (A), false, (B), (short)0, (C), false, false)

#define DIM      512
#define NTOK     1024          // 32*32 tokens per batch image
#define BATCH    8
#define ROWS     8192          // BATCH * NTOK
#define HEADS    8
#define DHEAD    64
#define F_QKV    1536

#if defined(__HIP_DEVICE_COMPILE__) && \
    __has_builtin(__builtin_amdgcn_global_load_async_to_lds_b128) && \
    __has_builtin(__builtin_amdgcn_s_wait_asynccnt)
#define HAVE_ASYNC_LDS 1
#else
#define HAVE_ASYNC_LDS 0
#endif

// 16B global -> LDS copy; async (ASYNCcnt) when the toolchain exposes it.
// Builtin signature (from hipcc diagnostic): arg0 = int4 addrspace(1)*,
// arg1 = int4 addrspace(3)*, then imm offset, imm cpol.
__device__ __forceinline__ void copy16_g2s(const bf16* g, bf16* s) {
#if HAVE_ASYNC_LDS
  __builtin_amdgcn_global_load_async_to_lds_b128(
      (__attribute__((address_space(1))) v4i*)(void*)g,
      (__attribute__((address_space(3))) v4i*)(void*)s, 0, 0);
#else
  *(uint4*)s = *(const uint4*)g;
#endif
}

__device__ __forceinline__ void wait_async_copies() {
#if HAVE_ASYNC_LDS
  __builtin_amdgcn_s_wait_asynccnt(0);
#endif
}

// ---------------------------------------------------------------------------
// Kernel 1: weights f32 -> bf16 hi/lo, transposed to [f][c] so that WMMA
// B-fragments (lane n = f, elements e over K = c) are contiguous 16B loads.
// ---------------------------------------------------------------------------
__global__ __launch_bounds__(256) void convert_weights(
    const float* __restrict__ wqkv, const float* __restrict__ wout,
    bf16* __restrict__ qT_hi, bf16* __restrict__ qT_lo,
    bf16* __restrict__ oT_hi, bf16* __restrict__ oT_lo)
{
  const int idx = blockIdx.x * 256 + threadIdx.x;
  const int N1 = F_QKV * DIM;     // 786432
  const int N2 = DIM * DIM;       // 262144
  if (idx < N1) {
    int f = idx / DIM, c = idx % DIM;
    float v = wqkv[c * F_QKV + f];
    bf16 h = (bf16)v;
    qT_hi[idx] = h;
    qT_lo[idx] = (bf16)(v - (float)h);
  } else if (idx < N1 + N2) {
    int j = idx - N1;
    int f = j / DIM, c = j % DIM;
    float v = wout[c * DIM + f];
    bf16 h = (bf16)v;
    oT_hi[j] = h;
    oT_lo[j] = (bf16)(v - (float)h);
  }
}

// ---------------------------------------------------------------------------
// Kernel 2: QKV projection. Block = 64x64 output tile, 8 waves:
// wave = rg(4 row groups of 16) x cg(2 col groups of 32 = 2 WMMA tiles).
// A fragments built on the fly from f32 x (hi + lo residual), B fragments
// loaded straight from the transposed bf16 weight copies. 3 WMMAs / tile / k.
// ---------------------------------------------------------------------------
__global__ __launch_bounds__(256) void qkv_gemm(
    const float* __restrict__ X,
    const bf16* __restrict__ WT_hi, const bf16* __restrict__ WT_lo,
    bf16* __restrict__ Qb, bf16* __restrict__ Kb, bf16* __restrict__ Vt)
{
  const int lane = threadIdx.x & 31;
  const int wave = threadIdx.x >> 5;
  const int half = lane >> 4;
  const int l16  = lane & 15;
  const int rg   = wave & 3;
  const int cg   = wave >> 2;
  const int m0   = blockIdx.x * 64 + rg * 16;
  const int n0   = blockIdx.y * 64 + cg * 32;

  const v8f zero8 = {0.f,0.f,0.f,0.f,0.f,0.f,0.f,0.f};
  v8f acc[2] = { zero8, zero8 };

  const float* xrow = X + (size_t)(m0 + l16) * DIM;

  for (int k0 = 0; k0 < DIM; k0 += 32) {
    // ---- A fragment (16x32 bf16): K = half*8 + {0..7} and half*8 + {16..23}
    const float* xp = xrow + k0 + half * 8;
    v8f xa = *(const v8f*)(xp);        // K = half*8 + 0..7
    v8f xb = *(const v8f*)(xp + 16);   // K = half*8 + 16..23
    V16 ah, al;
#pragma unroll
    for (int e = 0; e < 8; ++e) {
      bf16 h0 = (bf16)xa[e];
      ah.e[e]     = h0;
      al.e[e]     = (bf16)(xa[e] - (float)h0);
      bf16 h1 = (bf16)xb[e];
      ah.e[e + 8] = h1;
      al.e[e + 8] = (bf16)(xb[e] - (float)h1);
    }

#pragma unroll
    for (int t = 0; t < 2; ++t) {
      const size_t wbase = (size_t)(n0 + t * 16 + l16) * DIM + k0 + half * 8;
      V16 bh, bl;
      bh.h[0] = *(const v8bf*)(WT_hi + wbase);
      bh.h[1] = *(const v8bf*)(WT_hi + wbase + 16);
      bl.h[0] = *(const v8bf*)(WT_lo + wbase);
      bl.h[1] = *(const v8bf*)(WT_lo + wbase + 16);
      acc[t] = WMMA_BF16(ah.v, bh.v, acc[t]);   // hi*hi
      acc[t] = WMMA_BF16(ah.v, bl.v, acc[t]);   // hi*lo
      acc[t] = WMMA_BF16(al.v, bh.v, acc[t]);   // lo*hi
    }
  }

  // ---- epilogue: scatter into Q / K / Vt (block column spans one head)
  const int fbase = blockIdx.y * 64;
  const int which = fbase >> 9;            // 0=q 1=k 2=v
  const int head  = (fbase >> 6) & 7;
#pragma unroll
  for (int t = 0; t < 2; ++t) {
    const int dd = cg * 32 + t * 16 + l16; // 0..63 within head
#pragma unroll
    for (int r = 0; r < 8; ++r) {
      const int token = blockIdx.x * 64 + rg * 16 + half * 8 + r;
      const int b = token >> 10, n = token & (NTOK - 1);
      const size_t bh = (size_t)(b * HEADS + head);
      const bf16 v = (bf16)acc[t][r];
      if (which == 0)      Qb[(bh * NTOK + n) * DHEAD + dd] = v;
      else if (which == 1) Kb[(bh * NTOK + n) * DHEAD + dd] = v;
      else                 Vt[(bh * DHEAD + dd) * NTOK + n] = v;
    }
  }
}

// ---------------------------------------------------------------------------
// Kernel 3: flash attention. grid = (b*h = 64, 1024/128). Each wave owns 16
// query rows. Per 64-key step, the block stages the shared K tile [64x64]
// and V^T tile [64x64] into LDS via async copies (issued once per block,
// not once per wave), then all 8 waves read WMMA fragments from LDS.
// P bounces through wave-private LDS to convert D-layout f32 -> A-layout bf16.
// ---------------------------------------------------------------------------
#define PPITCH 72   // halves; 144B row stride -> conflict-free b128 reads

__global__ __launch_bounds__(256) void attention(
    const bf16* __restrict__ Qb, const bf16* __restrict__ Kb,
    const bf16* __restrict__ Vt, bf16* __restrict__ AO)
{
  __shared__ __align__(16) bf16 Ktile[64][PPITCH];   // [key j][d]
  __shared__ __align__(16) bf16 Vtile[64][PPITCH];   // [d][key j]
  __shared__ __align__(16) bf16 plds[8][16][PPITCH]; // per-wave P bounce

  const int tid  = threadIdx.x;
  const int lane = tid & 31;
  const int wave = tid >> 5;
  const int half = lane >> 4;
  const int l16  = lane & 15;
  const int bh   = blockIdx.x;                 // 0..63
  const int q0   = blockIdx.y * 128 + wave * 16;
  const size_t baseBH = (size_t)bh * NTOK * DHEAD;

  // Q fragments: A 16x32, two K-segments over d=0..63 (live whole kernel)
  V16 qa[2];
  const bf16* qrow = Qb + baseBH + (size_t)(q0 + l16) * DHEAD;
#pragma unroll
  for (int s = 0; s < 2; ++s) {
    qa[s].h[0] = *(const v8bf*)(qrow + s * 32 + half * 16);
    qa[s].h[1] = *(const v8bf*)(qrow + s * 32 + half * 16 + 8);
  }

  const v8f zero8 = {0.f,0.f,0.f,0.f,0.f,0.f,0.f,0.f};
  v8f o[4] = { zero8, zero8, zero8, zero8 };
  float mrow[8], lrow[8];
#pragma unroll
  for (int r = 0; r < 8; ++r) { mrow[r] = -1e30f; lrow[r] = 0.f; }

  for (int j0 = 0; j0 < NTOK; j0 += 64) {
    // ---- stage K[j0..j0+63][0..63] and Vt[0..63][j0..j0+63] into LDS.
    // 512 x 16B chunks per tile, 2 per thread; async -> ASYNCcnt.
#pragma unroll
    for (int c = 0; c < 2; ++c) {
      const int idx  = tid * 2 + c;        // 0..511
      const int row  = idx >> 3;           // 0..63
      const int col8 = (idx & 7) * 8;      // half offset within 64-wide row
      copy16_g2s(Kb + baseBH + (size_t)(j0 + row) * DHEAD + col8,
                 &Ktile[row][col8]);
      copy16_g2s(Vt + baseBH + (size_t)row * NTOK + j0 + col8,
                 &Vtile[row][col8]);
    }
    wait_async_copies();
    __syncthreads();

    // ---- S = (Q K^T) * scale, four 16x16 tiles over this 64-key block
    v8f s[4];
#pragma unroll
    for (int jt = 0; jt < 4; ++jt) {
      const bf16* krow = &Ktile[jt * 16 + l16][0];
      V16 kb0, kb1;
      kb0.h[0] = *(const v8bf*)(krow + half * 16);
      kb0.h[1] = *(const v8bf*)(krow + half * 16 + 8);
      kb1.h[0] = *(const v8bf*)(krow + 32 + half * 16);
      kb1.h[1] = *(const v8bf*)(krow + 32 + half * 16 + 8);
      v8f st = WMMA_BF16(qa[0].v, kb0.v, zero8);
      st     = WMMA_BF16(qa[1].v, kb1.v, st);
      s[jt] = st * 0.125f;                    // scale = 1/sqrt(64)
    }

    // ---- online softmax: row max over 64 cols (4 tiles + 16-lane xor tree)
    float mt[8];
#pragma unroll
    for (int r = 0; r < 8; ++r) {
      float v = fmaxf(fmaxf(s[0][r], s[1][r]), fmaxf(s[2][r], s[3][r]));
#pragma unroll
      for (int off = 1; off < 16; off <<= 1)
        v = fmaxf(v, __shfl_xor(v, off, 32));
      mt[r] = v;
    }
    float alpha[8];
#pragma unroll
    for (int r = 0; r < 8; ++r) {
      const float mn = fmaxf(mrow[r], mt[r]);
      alpha[r] = __expf(mrow[r] - mn);
      mrow[r]  = mn;
    }
    float rs[8];
#pragma unroll
    for (int r = 0; r < 8; ++r) rs[r] = 0.f;
#pragma unroll
    for (int jt = 0; jt < 4; ++jt)
#pragma unroll
      for (int r = 0; r < 8; ++r) {
        const float p = __expf(s[jt][r] - mrow[r]);
        s[jt][r] = p;
        rs[r] += p;
      }
#pragma unroll
    for (int r = 0; r < 8; ++r) {
      float v = rs[r];
#pragma unroll
      for (int off = 1; off < 16; off <<= 1)
        v += __shfl_xor(v, off, 32);
      lrow[r] = lrow[r] * alpha[r] + v;
    }
#pragma unroll
    for (int nt = 0; nt < 4; ++nt)
#pragma unroll
      for (int r = 0; r < 8; ++r)
        o[nt][r] *= alpha[r];

    // ---- P: D-layout f32 -> LDS -> A-layout bf16 (wave-private bounce)
    __builtin_amdgcn_wave_barrier();
#pragma unroll
    for (int jt = 0; jt < 4; ++jt)
#pragma unroll
      for (int r = 0; r < 8; ++r)
        plds[wave][half * 8 + r][jt * 16 + l16] = (bf16)s[jt][r];
    __builtin_amdgcn_wave_barrier();

    // ---- O += P * V (V fragments from the staged LDS tile)
#pragma unroll
    for (int js = 0; js < 2; ++js) {
      V16 pa;
      const bf16* pr = &plds[wave][l16][js * 32 + half * 16];
      pa.h[0] = *(const v8bf*)(pr);
      pa.h[1] = *(const v8bf*)(pr + 8);
#pragma unroll
      for (int nt = 0; nt < 4; ++nt) {
        const bf16* vr = &Vtile[nt * 16 + l16][js * 32 + half * 16];
        V16 vb;
        vb.h[0] = *(const v8bf*)(vr);
        vb.h[1] = *(const v8bf*)(vr + 8);
        o[nt] = WMMA_BF16(pa.v, vb.v, o[nt]);
      }
    }
    __syncthreads();   // all waves done with K/V tiles before restaging
  }

  // ---- epilogue: normalize and write AO[b, n, h*64 + d] (bf16)
  const int b = bh >> 3, head = bh & 7;
#pragma unroll
  for (int nt = 0; nt < 4; ++nt)
#pragma unroll
    for (int r = 0; r < 8; ++r) {
      const int q = q0 + half * 8 + r;
      const float val = o[nt][r] / lrow[r];
      AO[((size_t)(b * NTOK + q)) * DIM + head * DHEAD + nt * 16 + l16] = (bf16)val;
    }
}

// ---------------------------------------------------------------------------
// Kernel 4: output projection + bias. A = AO (plain bf16), B = w_out split
// hi/lo -> 2 WMMAs per tile per k-step; f32 result streamed to d_out with
// non-temporal stores (written once, never re-read).
// ---------------------------------------------------------------------------
__global__ __launch_bounds__(256) void out_gemm(
    const bf16* __restrict__ AO,
    const bf16* __restrict__ WT_hi, const bf16* __restrict__ WT_lo,
    const float* __restrict__ bias, float* __restrict__ out)
{
  const int lane = threadIdx.x & 31;
  const int wave = threadIdx.x >> 5;
  const int half = lane >> 4;
  const int l16  = lane & 15;
  const int rg   = wave & 3;
  const int cg   = wave >> 2;
  const int m0   = blockIdx.x * 64 + rg * 16;
  const int n0   = blockIdx.y * 64 + cg * 32;

  const v8f zero8 = {0.f,0.f,0.f,0.f,0.f,0.f,0.f,0.f};
  v8f acc[2] = { zero8, zero8 };

  const bf16* arow = AO + (size_t)(m0 + l16) * DIM;

  for (int k0 = 0; k0 < DIM; k0 += 32) {
    V16 a;
    a.h[0] = *(const v8bf*)(arow + k0 + half * 16);
    a.h[1] = *(const v8bf*)(arow + k0 + half * 16 + 8);
#pragma unroll
    for (int t = 0; t < 2; ++t) {
      const size_t wbase = (size_t)(n0 + t * 16 + l16) * DIM + k0 + half * 16;
      V16 bh, bl;
      bh.h[0] = *(const v8bf*)(WT_hi + wbase);
      bh.h[1] = *(const v8bf*)(WT_hi + wbase + 8);
      bl.h[0] = *(const v8bf*)(WT_lo + wbase);
      bl.h[1] = *(const v8bf*)(WT_lo + wbase + 8);
      acc[t] = WMMA_BF16(a.v, bh.v, acc[t]);
      acc[t] = WMMA_BF16(a.v, bl.v, acc[t]);
    }
  }

#pragma unroll
  for (int t = 0; t < 2; ++t) {
    const int f = n0 + t * 16 + l16;
    const float bv = bias[f];
#pragma unroll
    for (int r = 0; r < 8; ++r) {
      const int token = blockIdx.x * 64 + rg * 16 + half * 8 + r;
      __builtin_nontemporal_store(acc[t][r] + bv, &out[(size_t)token * DIM + f]);
    }
  }
}

// ---------------------------------------------------------------------------
extern "C" void kernel_launch(void* const* d_in, const int* in_sizes, int n_in,
                              void* d_out, int out_size, void* d_ws, size_t ws_size,
                              hipStream_t stream) {
  const float* x     = (const float*)d_in[0];   // [8,32,32,512]
  const float* w_qkv = (const float*)d_in[1];   // [512,1536]
  const float* w_out = (const float*)d_in[2];   // [512,512]
  const float* b_out = (const float*)d_in[3];   // [512]
  float* out = (float*)d_out;                   // [8,32,32,512] f32

  // workspace carve (bf16 halves), total ~36 MB
  bf16* p = (bf16*)d_ws;
  bf16* wqkvT_hi = p;                 p += (size_t)F_QKV * DIM;
  bf16* wqkvT_lo = p;                 p += (size_t)F_QKV * DIM;
  bf16* woutT_hi = p;                 p += (size_t)DIM * DIM;
  bf16* woutT_lo = p;                 p += (size_t)DIM * DIM;
  bf16* Qb = p;                       p += (size_t)BATCH * HEADS * NTOK * DHEAD;
  bf16* Kb = p;                       p += (size_t)BATCH * HEADS * NTOK * DHEAD;
  bf16* Vt = p;                       p += (size_t)BATCH * HEADS * NTOK * DHEAD;
  bf16* AO = p;                       p += (size_t)ROWS * DIM;

  convert_weights<<<4096, 256, 0, stream>>>(w_qkv, w_out,
                                            wqkvT_hi, wqkvT_lo, woutT_hi, woutT_lo);
  qkv_gemm<<<dim3(ROWS / 64, F_QKV / 64), 256, 0, stream>>>(
      x, wqkvT_hi, wqkvT_lo, Qb, Kb, Vt);
  attention<<<dim3(BATCH * HEADS, NTOK / 128), 256, 0, stream>>>(Qb, Kb, Vt, AO);
  out_gemm<<<dim3(ROWS / 64, DIM / 64), 256, 0, stream>>>(
      AO, woutT_hi, woutT_lo, b_out, out);
}
